// MeshHeadModule_6614249636235
// MI455X (gfx1250) — matrix operand order repaired and checked by reading, not compiled, not run on hardware
//
#include <hip/hip_runtime.h>
#include <hip/hip_fp16.h>
#include <math.h>

// ---------------------------------------------------------------------------
// CDNA5 (gfx1250) WMMA types
// ---------------------------------------------------------------------------
typedef __attribute__((ext_vector_type(16))) _Float16 v16h;
typedef __attribute__((ext_vector_type(8)))  _Float16 v8h;
typedef __attribute__((ext_vector_type(8)))  float    v8f;

union V16 { v16h v; v8h h[2]; };

// ---------------------------------------------------------------------------
// Problem constants (from reference)
// ---------------------------------------------------------------------------
constexpr int   Bc    = 2;
constexpr int   Nc    = 131072;
constexpr int   FEATc = 128;
constexpr int   CODEc = 256;
constexpr int   LMc   = 68;
constexpr float NEARc = 0.1f, FARc = 0.25f, DSCALE = 0.3f;

constexpr int TPB = 64;    // 2 waves of 32
constexpr int VPB = 64;    // vertices per block (32 per wave: two 16-row M-tiles)
constexpr int LD  = 384;   // X buffer row stride (halves)
constexpr int HOFF = 128;  // hidden activations live at X cols [128, 128+H)

// Swizzled weight offsets in d_ws (halves). tile = 512 halves (32 lanes x 16).
constexpr size_t OFF_SC1 = 0;                              // 12x16 tiles
constexpr size_t OFF_SC2 = OFF_SC1 + (size_t)12*16*512;    //  8x1
constexpr size_t OFF_PC1 = OFF_SC2 + (size_t) 8* 1*512;    //  8x8
constexpr size_t OFF_PC2 = OFF_PC1 + (size_t) 8* 8*512;    //  4x1
constexpr size_t OFF_EC1 = OFF_PC2 + (size_t) 4* 1*512;    //  6x8
constexpr size_t OFF_EC2 = OFF_EC1 + (size_t) 6* 8*512;    //  4x1
constexpr size_t OFF_SD1 = OFF_EC2 + (size_t) 4* 1*512;    // 10x16
constexpr size_t OFF_SD2 = OFF_SD1 + (size_t)10*16*512;    //  8x1
constexpr size_t OFF_PD1 = OFF_SD2 + (size_t) 8* 1*512;    //  6x8
constexpr size_t OFF_PD2 = OFF_PD1 + (size_t) 6* 8*512;    //  4x1
constexpr size_t OFF_ED1 = OFF_PD2 + (size_t) 4* 1*512;    //  4x8
constexpr size_t OFF_ED2 = OFF_ED1 + (size_t) 4* 8*512;    //  4x1

// ---------------------------------------------------------------------------
// Prep kernel: f32 weights -> f16, swizzled into this file's WMMA B layout (contiguous k, matching the A loads).
// B 32x16 tile, lane l: column N = l&15, K = (l>>4)*16 + h  (h = half 0..15),
// stored as 16 contiguous halves per lane -> 32B coalesced per-lane loads.
// Row remap: X col c -> weight row: c < k_lo ? c
//            : (code_dst <= c < code_dst+code_len) ? k_lo + (c - code_dst) : 0
// ---------------------------------------------------------------------------
__global__ void prep_swizzle(const float* __restrict__ src, _Float16* __restrict__ dst,
                             int Nact, int KT, int NT,
                             int k_lo, int code_dst, int code_len)
{
    int id = blockIdx.x * 256 + threadIdx.x;
    int total = KT * NT * 512;
    if (id >= total) return;
    int tile = id >> 9;
    int rem  = id & 511;
    int lane = rem >> 4, h = rem & 15;
    int j = tile / KT, s = tile - j * KT;           // tile index = j*KT + s (n-major)
    int kcol = s * 32 + ((lane >> 4) << 4) + h;     // X column (K index in padded space)
    int n    = (j << 4) + (lane & 15);              // output column
    int srow;
    if (kcol < k_lo)                                   srow = kcol;
    else if (kcol >= code_dst && kcol < code_dst + code_len) srow = k_lo + (kcol - code_dst);
    else                                               srow = -1;
    float v = 0.f;
    if (srow >= 0 && n < Nact) v = src[(size_t)srow * Nact + n];
    dst[id] = (_Float16)v;
}

// ---------------------------------------------------------------------------
// One 32-vertex 2-layer MLP on a wave (two 16-row M-tiles share each B tile:
// 2 WMMAs per 1KB B load -> 32 FLOP/byte on the weight stream).
//   layer1: [32 x K1] @ [K1 x H] + b1, leaky_relu(0.2)  (WMMA f16->f32)
//   layer2: [32 x H ] @ [H x 16] + b2 (cols 0..2 valid) -> out3 (pre-act)
// X row-major in LDS, stride LD halves. Hidden written into X cols [HOFF, HOFF+H).
// A fragment layout (16-bit A 16x32): lane M=l&15, lg=l>>4:
//   halves 0..15 = K kbase+16*lg .. +15 (matches B swizzle)
// ---------------------------------------------------------------------------
template<int K1, int H>
__device__ __forceinline__ void mlp32(_Float16* Xw, float* out3w,
                                      const _Float16* __restrict__ W1,
                                      const float*    __restrict__ b1,
                                      const _Float16* __restrict__ W2,
                                      const float*    __restrict__ b2,
                                      int lane)
{
    constexpr int KT  = K1 / 32;
    constexpr int NT  = H  / 16;
    constexpr int KT2 = H  / 32;
    const int M  = lane & 15;
    const int lg = lane >> 4;

    // Hoist all A fragments for both M-tiles (frees X for hidden writes).
    V16 A0[KT], A1[KT];
#pragma unroll
    for (int s = 0; s < KT; ++s) {
        const _Float16* p0 = Xw + M * LD + s * 32 + 16 * lg;
        A0[s].h[0] = *(const v8h*)p0;
        A0[s].h[1] = *(const v8h*)(p0 + 8);
        const _Float16* p1 = p0 + 16 * LD;
        A1[s].h[0] = *(const v8h*)p1;
        A1[s].h[1] = *(const v8h*)(p1 + 8);
    }

#pragma unroll
    for (int j = 0; j < NT; ++j) {
        v8f acc0 = {0.f,0.f,0.f,0.f,0.f,0.f,0.f,0.f};
        v8f acc1 = {0.f,0.f,0.f,0.f,0.f,0.f,0.f,0.f};
#pragma unroll
        for (int s = 0; s < KT; ++s) {
            V16 Bf;
            Bf.v = *(const v16h*)(W1 + (((size_t)(j * KT + s)) << 9) + ((size_t)lane << 4));
            acc0 = __builtin_amdgcn_wmma_f32_16x16x32_f16(false, A0[s].v, false, Bf.v,
                                                          (short)0, acc0, false, false);
            acc1 = __builtin_amdgcn_wmma_f32_16x16x32_f16(false, A1[s].v, false, Bf.v,
                                                          (short)0, acc1, false, false);
        }
        const int   n    = (j << 4) + M;
        const float bias = b1[n];
#pragma unroll
        for (int r = 0; r < 8; ++r) {
            float x0 = acc0[r] + bias;
            x0 = x0 > 0.f ? x0 : 0.2f * x0;             // leaky_relu(0.2)
            Xw[(r + 8 * lg) * LD + HOFF + n] = (_Float16)x0;
            float x1 = acc1[r] + bias;
            x1 = x1 > 0.f ? x1 : 0.2f * x1;
            Xw[(16 + r + 8 * lg) * LD + HOFF + n] = (_Float16)x1;
        }
    }
    __asm__ volatile("s_wait_dscnt 0x0" ::: "memory");  // layer1 stores -> layer2 loads

    v8f acc0 = {0.f,0.f,0.f,0.f,0.f,0.f,0.f,0.f};
    v8f acc1 = {0.f,0.f,0.f,0.f,0.f,0.f,0.f,0.f};
#pragma unroll
    for (int s = 0; s < KT2; ++s) {
        const _Float16* p0 = Xw + M * LD + HOFF + s * 32 + 16 * lg;
        V16 Af0; Af0.h[0] = *(const v8h*)p0;            Af0.h[1] = *(const v8h*)(p0 + 8);
        const _Float16* p1 = p0 + 16 * LD;
        V16 Af1; Af1.h[0] = *(const v8h*)p1;            Af1.h[1] = *(const v8h*)(p1 + 8);
        V16 Bf;  Bf.v = *(const v16h*)(W2 + (((size_t)s) << 9) + ((size_t)lane << 4));
        acc0 = __builtin_amdgcn_wmma_f32_16x16x32_f16(false, Af0.v, false, Bf.v,
                                                      (short)0, acc0, false, false);
        acc1 = __builtin_amdgcn_wmma_f32_16x16x32_f16(false, Af1.v, false, Bf.v,
                                                      (short)0, acc1, false, false);
    }
    if (M < 3) {
        const float bias = b2[M];
#pragma unroll
        for (int r = 0; r < 8; ++r) {
            out3w[(r + 8 * lg) * 4 + M]        = acc0[r] + bias;   // pre-activation
            out3w[(16 + r + 8 * lg) * 4 + M]   = acc1[r] + bias;
        }
    }
    __asm__ volatile("s_wait_dscnt 0x0" ::: "memory");
}

// ---------------------------------------------------------------------------
// Main fused kernel: one block = 64 vertices (2 waves x 32-row WMMA tiles).
// Every thread owns exactly one vertex for the scalar phases.
// ---------------------------------------------------------------------------
__global__ __launch_bounds__(TPB) void mesh_head_main(
    const float* __restrict__ verts_g, const float* __restrict__ feat_g,
    const float* __restrict__ lm_g,    const float* __restrict__ scode_g,
    const float* __restrict__ pose_g,  const float* __restrict__ py_g,
    const float* __restrict__ scale_g,
    const float* __restrict__ scB1, const float* __restrict__ scB2,
    const float* __restrict__ pcB1, const float* __restrict__ pcB2,
    const float* __restrict__ ecB1, const float* __restrict__ ecB2,
    const float* __restrict__ sdB1, const float* __restrict__ sdB2,
    const float* __restrict__ pdB1, const float* __restrict__ pdB2,
    const float* __restrict__ edB1, const float* __restrict__ edB2,
    const _Float16* __restrict__ ws, float* __restrict__ out_g)
{
    __shared__ __attribute__((aligned(16))) _Float16 Xs[VPB * LD];
    __shared__ __attribute__((aligned(16))) _Float16 codeS[CODEc];
    __shared__ __attribute__((aligned(16))) _Float16 embP[128];   // pose emb 102 + pad
    __shared__ __attribute__((aligned(16))) _Float16 embG[64];    // gaze emb 34 + pad
    __shared__ float out3[VPB * 4];
    __shared__ float lmS[LMc * 3];
    __shared__ float swS[VPB], ewS[VPB];
    __shared__ float vxS[VPB], vyS[VPB], vzS[VPB];

    const int t    = (int)threadIdx.x;
    const int lane = t & 31;
    const int wv   = t >> 5;                 // 0..1
    const int v0   = (int)blockIdx.x * VPB;
    const int b    = v0 / Nc;
    _Float16* Xw    = Xs   + (wv * 32) * LD;
    float*    out3w = out3 + (wv * 32) * 4;

    // ---- per-batch staging ------------------------------------------------
    for (int i = t; i < LMc * 3; i += TPB) lmS[i] = lm_g[b * LMc * 3 + i];
    for (int i = t; i < CODEc;   i += TPB) codeS[i] = (_Float16)scode_g[b * CODEc + i];
    for (int i = t; i < 128; i += TPB) {
        float val = 0.f;
        if (i < 102) {
            if (i < 6) val = pose_g[b * 6 + i];
            else {
                int j = i - 6, k = j / 12, r = j % 12;
                float f = (float)(1 << k);
                float p = pose_g[b * 6 + (r % 6)];
                val = (r < 6) ? __sinf(f * p) : __cosf(f * p);
            }
        }
        embP[i] = (_Float16)val;
    }
    if (t < 64) {
        int i = t;
        float val = 0.f;
        if (i < 34) {
            if (i < 2) val = py_g[b * 2 + i];
            else {
                int j = i - 2, k = j / 4, r = j % 4;
                float f = (float)(1 << k);
                float p = py_g[b * 2 + (r % 2)];
                val = (r < 2) ? __sinf(f * p) : __cosf(f * p);
            }
        }
        embG[i] = (_Float16)val;
    }
    {
        int g = v0 + t;
        vxS[t] = verts_g[g * 3 + 0];
        vyS[t] = verts_g[g * 3 + 1];
        vzS[t] = verts_g[g * 3 + 2];
    }
    __syncthreads();

    // ---- landmark distance weights (on squared distances, like reference) --
    {
        float x = vxS[t], y = vyS[t], z = vzS[t];
        float dmin = 1e30f, emin = 1e30f;
#pragma unroll 4
        for (int l = 0; l < LMc; ++l) {
            float dx = x - lmS[l * 3 + 0];
            float dy = y - lmS[l * 3 + 1];
            float dz = z - lmS[l * 3 + 2];
            float d2 = dx * dx + dy * dy + dz * dz;
            dmin = fminf(dmin, d2);
            if (l >= 36 && l < 48) emin = fminf(emin, d2);
        }
        const float inv = 1.f / (FARc - NEARc);
        swS[t] = fminf(fmaxf((FARc - dmin) * inv, 0.f), 1.f);
        ewS[t] = fminf(fmaxf((FARc - emin) * inv, 0.f), 1.f);
    }

    // ---- fill X: features (cols 0..127, kept across all 3 color MLPs) -----
    for (int i = t; i < VPB * 32; i += TPB) {
        int row = i >> 5, c4 = i & 31;
        float4 f = ((const float4*)(feat_g + (size_t)(v0 + row) * FEATc))[c4];
        _Float16* d = Xs + row * LD + c4 * 4;
        d[0] = (_Float16)f.x; d[1] = (_Float16)f.y;
        d[2] = (_Float16)f.z; d[3] = (_Float16)f.w;
    }
    // shape code -> cols 128..383
    for (int i = t; i < VPB * 32; i += TPB) {
        int row = i >> 5, c = i & 31;
        *(v8h*)(Xs + row * LD + 128 + c * 8) = ((const v8h*)codeS)[c];
    }
    __syncthreads();

    float c0 = 0.f, c1 = 0.f, c2 = 0.f;

    // ---- shape color: [384 -> 256 -> 3], sigmoid * shape_w ----------------
    mlp32<384, 256>(Xw, out3w, ws + OFF_SC1, scB1, ws + OFF_SC2, scB2, lane);
    __syncthreads();
    {
        float w = swS[t];
        c0 = w / (1.f + __expf(-out3[t * 4 + 0]));
        c1 = w / (1.f + __expf(-out3[t * 4 + 1]));
        c2 = w / (1.f + __expf(-out3[t * 4 + 2]));
    }
    // pose emb -> cols 128..255
    for (int i = t; i < VPB * 16; i += TPB) {
        int row = i >> 4, c = i & 15;
        *(v8h*)(Xs + row * LD + 128 + c * 8) = ((const v8h*)embP)[c];
    }
    __syncthreads();

    // ---- pose color: [230(pad 256) -> 128 -> 3], sigmoid * (1-shape_w) ----
    mlp32<256, 128>(Xw, out3w, ws + OFF_PC1, pcB1, ws + OFF_PC2, pcB2, lane);
    __syncthreads();
    {
        float w = 1.f - swS[t];
        c0 += w / (1.f + __expf(-out3[t * 4 + 0]));
        c1 += w / (1.f + __expf(-out3[t * 4 + 1]));
        c2 += w / (1.f + __expf(-out3[t * 4 + 2]));
    }
    // gaze emb -> cols 128..191
    for (int i = t; i < VPB * 8; i += TPB) {
        int row = i >> 3, c = i & 7;
        *(v8h*)(Xs + row * LD + 128 + c * 8) = ((const v8h*)embG)[c];
    }
    __syncthreads();

    // ---- eye color: [162(pad 192) -> 128 -> 3], sigmoid * eye_w -----------
    mlp32<192, 128>(Xw, out3w, ws + OFF_EC1, ecB1, ws + OFF_EC2, ecB2, lane);
    __syncthreads();
    {
        float w = ewS[t];
        c0 += w / (1.f + __expf(-out3[t * 4 + 0]));
        c1 += w / (1.f + __expf(-out3[t * 4 + 1]));
        c2 += w / (1.f + __expf(-out3[t * 4 + 2]));
    }

    // ======================= deform stages =================================
    // X layout: cols 0..50 posemb(vert), 51..63 zero, code at col 64 (16B aligned;
    // prep kernel remapped weight rows to match).

    // ---- stage 1: shape deform [307 -> pad 320 -> 256 -> 3] ---------------
    {
        _Float16* row = Xs + t * LD;
        float p0 = vxS[t], p1 = vyS[t], p2 = vzS[t];
        row[0] = (_Float16)p0; row[1] = (_Float16)p1; row[2] = (_Float16)p2;
#pragma unroll
        for (int k = 0; k < 8; ++k) {
            float f = (float)(1 << k);
            row[3 + k * 6 + 0] = (_Float16)__sinf(f * p0);
            row[3 + k * 6 + 1] = (_Float16)__sinf(f * p1);
            row[3 + k * 6 + 2] = (_Float16)__sinf(f * p2);
            row[3 + k * 6 + 3] = (_Float16)__cosf(f * p0);
            row[3 + k * 6 + 4] = (_Float16)__cosf(f * p1);
            row[3 + k * 6 + 5] = (_Float16)__cosf(f * p2);
        }
        for (int c = 51; c < 64; ++c) row[c] = (_Float16)0.f;
    }
    for (int i = t; i < VPB * 32; i += TPB) {   // code -> cols 64..319
        int row = i >> 5, c = i & 31;
        *(v8h*)(Xs + row * LD + 64 + c * 8) = ((const v8h*)codeS)[c];
    }
    __syncthreads();
    mlp32<320, 256>(Xw, out3w, ws + OFF_SD1, sdB1, ws + OFF_SD2, sdB2, lane);
    __syncthreads();
    {
        float w = swS[t] * DSCALE;
        vxS[t] += w * tanhf(out3[t * 4 + 0]);
        vyS[t] += w * tanhf(out3[t * 4 + 1]);
        vzS[t] += w * tanhf(out3[t * 4 + 2]);
    }

    // ---- stage 2: pose deform [153 -> pad 192 -> 128 -> 3] ----------------
    {
        _Float16* row = Xs + t * LD;
        float p0 = vxS[t], p1 = vyS[t], p2 = vzS[t];
        row[0] = (_Float16)p0; row[1] = (_Float16)p1; row[2] = (_Float16)p2;
#pragma unroll
        for (int k = 0; k < 8; ++k) {
            float f = (float)(1 << k);
            row[3 + k * 6 + 0] = (_Float16)__sinf(f * p0);
            row[3 + k * 6 + 1] = (_Float16)__sinf(f * p1);
            row[3 + k * 6 + 2] = (_Float16)__sinf(f * p2);
            row[3 + k * 6 + 3] = (_Float16)__cosf(f * p0);
            row[3 + k * 6 + 4] = (_Float16)__cosf(f * p1);
            row[3 + k * 6 + 5] = (_Float16)__cosf(f * p2);
        }
    }
    for (int i = t; i < VPB * 16; i += TPB) {   // pose emb -> cols 64..191
        int row = i >> 4, c = i & 15;
        *(v8h*)(Xs + row * LD + 64 + c * 8) = ((const v8h*)embP)[c];
    }
    __syncthreads();
    mlp32<192, 128>(Xw, out3w, ws + OFF_PD1, pdB1, ws + OFF_PD2, pdB2, lane);
    __syncthreads();
    {
        float w = (1.f - swS[t]) * DSCALE;
        vxS[t] += w * tanhf(out3[t * 4 + 0]);
        vyS[t] += w * tanhf(out3[t * 4 + 1]);
        vzS[t] += w * tanhf(out3[t * 4 + 2]);
    }

    // ---- stage 3: eye deform [85 -> pad 128 -> 128 -> 3] ------------------
    {
        _Float16* row = Xs + t * LD;
        float p0 = vxS[t], p1 = vyS[t], p2 = vzS[t];
        row[0] = (_Float16)p0; row[1] = (_Float16)p1; row[2] = (_Float16)p2;
#pragma unroll
        for (int k = 0; k < 8; ++k) {
            float f = (float)(1 << k);
            row[3 + k * 6 + 0] = (_Float16)__sinf(f * p0);
            row[3 + k * 6 + 1] = (_Float16)__sinf(f * p1);
            row[3 + k * 6 + 2] = (_Float16)__sinf(f * p2);
            row[3 + k * 6 + 3] = (_Float16)__cosf(f * p0);
            row[3 + k * 6 + 4] = (_Float16)__cosf(f * p1);
            row[3 + k * 6 + 5] = (_Float16)__cosf(f * p2);
        }
    }
    for (int i = t; i < VPB * 8; i += TPB) {    // gaze emb -> cols 64..127
        int row = i >> 3, c = i & 7;
        *(v8h*)(Xs + row * LD + 64 + c * 8) = ((const v8h*)embG)[c];
    }
    __syncthreads();
    mlp32<128, 128>(Xw, out3w, ws + OFF_ED1, edB1, ws + OFF_ED2, edB2, lane);
    __syncthreads();
    {
        float w = ewS[t] * DSCALE;
        float vx = vxS[t] + w * tanhf(out3[t * 4 + 0]);
        float vy = vyS[t] + w * tanhf(out3[t * 4 + 1]);
        float vz = vzS[t] + w * tanhf(out3[t * 4 + 2]);

        // ---- rigid transform: out = R @ (v * S) + T -----------------------
        float wx = pose_g[b * 6 + 0], wy = pose_g[b * 6 + 1], wz = pose_g[b * 6 + 2];
        float th = fmaxf(sqrtf(wx * wx + wy * wy + wz * wz), 1e-8f);
        float kx = wx / th, ky = wy / th, kz = wz / th;
        float st = __sinf(th), ct = __cosf(th), vc = 1.f - ct;
        float r00 = ct + vc * kx * kx, r01 = -st * kz + vc * kx * ky, r02 =  st * ky + vc * kx * kz;
        float r10 =  st * kz + vc * ky * kx, r11 = ct + vc * ky * ky, r12 = -st * kx + vc * ky * kz;
        float r20 = -st * ky + vc * kz * kx, r21 =  st * kx + vc * kz * ky, r22 = ct + vc * kz * kz;
        float S = scale_g[b];
        float x = vx * S, y = vy * S, z = vz * S;
        float ox = r00 * x + r01 * y + r02 * z + pose_g[b * 6 + 3];
        float oy = r10 * x + r11 * y + r12 * z + pose_g[b * 6 + 4];
        float oz = r20 * x + r21 * y + r22 * z + pose_g[b * 6 + 5];

        size_t g = (size_t)(v0 + t) * 6;
        out_g[g + 0] = ox; out_g[g + 1] = oy; out_g[g + 2] = oz;
        out_g[g + 3] = c0; out_g[g + 4] = c1; out_g[g + 5] = c2;
    }
}

// ---------------------------------------------------------------------------
// Launch: 12 weight-swizzle preps (weights ~0.5MB, L2-resident), then the
// fused WMMA kernel over all B*N vertices.
// d_in order (pytree leaves): 0 verts, 1 feats, 2 landmarks, 3 shape_code,
// 4 pose, 5 pitchyaw, 6 scale, then 6 MLPs x (W1,b1,W2,b2) = indices 7..30.
// ---------------------------------------------------------------------------
extern "C" void kernel_launch(void* const* d_in, const int* in_sizes, int n_in,
                              void* d_out, int out_size, void* d_ws, size_t ws_size,
                              hipStream_t stream)
{
    (void)in_sizes; (void)n_in; (void)out_size; (void)ws_size;
    _Float16* ws = (_Float16*)d_ws;

    struct PD { int idx; size_t off; int Nact, KT, NT, k_lo, code_dst, code_len; };
    const PD pds[12] = {
        { 7, OFF_SC1, 256, 12, 16, 128, 128, 256 },   // shape color W1 (384x256)
        { 9, OFF_SC2,   3,  8,  1, 256, 256,   0 },   // shape color W2 (256x3)
        {11, OFF_PC1, 128,  8,  8, 128, 128, 102 },   // pose  color W1 (230x128)
        {13, OFF_PC2,   3,  4,  1, 128, 128,   0 },   // pose  color W2 (128x3)
        {15, OFF_EC1, 128,  6,  8, 128, 128,  34 },   // eye   color W1 (162x128)
        {17, OFF_EC2,   3,  4,  1, 128, 128,   0 },   // eye   color W2 (128x3)
        {19, OFF_SD1, 256, 10, 16,  51,  64, 256 },   // shape deform W1 (307x256)
        {21, OFF_SD2,   3,  8,  1, 256, 256,   0 },   // shape deform W2 (256x3)
        {23, OFF_PD1, 128,  6,  8,  51,  64, 102 },   // pose  deform W1 (153x128)
        {25, OFF_PD2,   3,  4,  1, 128, 128,   0 },   // pose  deform W2 (128x3)
        {27, OFF_ED1, 128,  4,  8,  51,  64,  34 },   // eye   deform W1 (85x128)
        {29, OFF_ED2,   3,  4,  1, 128, 128,   0 },   // eye   deform W2 (128x3)
    };
    for (int i = 0; i < 12; ++i) {
        int total = pds[i].KT * pds[i].NT * 512;
        prep_swizzle<<<(total + 255) / 256, 256, 0, stream>>>(
            (const float*)d_in[pds[i].idx], ws + pds[i].off,
            pds[i].Nact, pds[i].KT, pds[i].NT,
            pds[i].k_lo, pds[i].code_dst, pds[i].code_len);
    }

    mesh_head_main<<<(Bc * Nc) / VPB, TPB, 0, stream>>>(
        (const float*)d_in[0],  (const float*)d_in[1],  (const float*)d_in[2],
        (const float*)d_in[3],  (const float*)d_in[4],  (const float*)d_in[5],
        (const float*)d_in[6],
        (const float*)d_in[8],  (const float*)d_in[10],
        (const float*)d_in[12], (const float*)d_in[14],
        (const float*)d_in[16], (const float*)d_in[18],
        (const float*)d_in[20], (const float*)d_in[22],
        (const float*)d_in[24], (const float*)d_in[26],
        (const float*)d_in[28], (const float*)d_in[30],
        ws, (float*)d_out);
}
